// HLN_5291399708951
// MI455X (gfx1250) — matrix operand order repaired and checked by reading, not compiled, not run on hardware
//
#include <hip/hip_runtime.h>
#include <hip/hip_bf16.h>

typedef __bf16 bf16;
typedef __attribute__((ext_vector_type(16))) __bf16 v16bf;
typedef __attribute__((ext_vector_type(8)))  float  v8f;

#define NB 1024
#define SSQ 200
#define PPASS 3
#define VV 50000
#define G3 192   // 3*H

// fast transcendentals: v_rcp_f32 / v_exp_f32 are TRANS ops that co-execute with WMMA
static __device__ __forceinline__ float frcp(float x)  { return __builtin_amdgcn_rcpf(x); }
static __device__ __forceinline__ float sigmf(float x) { return frcp(1.0f + __expf(-x)); }
static __device__ __forceinline__ float tanhf_fast(float x) {
    return 2.0f * frcp(1.0f + __expf(-2.0f * x)) - 1.0f;
}

// A fragment (16x32 bf16, M x K) from row-major [16][64] at K-offset kk*32.
static __device__ __forceinline__ v16bf afrag(const bf16* __restrict__ A, int lane, int kk) {
    const int m = lane & 15, half = lane >> 4;
    const bf16* p = A + m * 64 + kk * 32 + (half ? 16 : 0);
    v16bf a;
#pragma unroll
    for (int e = 0; e < 8; ++e) { a[e] = p[e]; a[8 + e] = p[8 + e]; }
    return a;
}

// B fragment (32x16 bf16, K x N) with B[k][n] = W[n][k], W row-major [16][64].
static __device__ __forceinline__ v16bf bfrag(const bf16* __restrict__ Wt, int lane, int kk) {
    const bf16* p = Wt + (lane & 15) * 64 + kk * 32 + (lane >> 4) * 16;
    v16bf b;
#pragma unroll
    for (int e = 0; e < 16; ++e) b[e] = p[e];
    return b;
}

static __device__ __forceinline__ v8f wmma_bf(v16bf a, v16bf b, v8f c) {
    return __builtin_amdgcn_wmma_f32_16x16x32_bf16(false, a, false, b, (short)0, c, false, false);
}

#define V8F_ZERO {0.f,0.f,0.f,0.f,0.f,0.f,0.f,0.f}

// ---------------- conversion kernels ----------------
__global__ void cvt_bf16_kernel(const float* __restrict__ src, bf16* __restrict__ dst, int n) {
    int i = blockIdx.x * blockDim.x + threadIdx.x;
    if (i < n) dst[i] = (bf16)src[i];
}

__global__ void padq_kernel(const float* __restrict__ wq, const float* __restrict__ uq,
                            bf16* __restrict__ wqp, bf16* __restrict__ uqp) {
    int i = blockIdx.x * blockDim.x + threadIdx.x;
    if (i < 16 * 64) {
        int r = i >> 6, c = i & 63;
        wqp[i] = (bf16)((r < 2) ? wq[r * 64 + c] : 0.0f);
        uqp[i] = (bf16)((r < 2) ? uq[r * 64 + c] : 0.0f);
    }
}

// ---------------- RNN scan: one WG = 16 batch rows, all 3 passes x 200 steps ----------------
// 128 threads = 4 waves. Wave w owns n-tiles {w, w+4, w+8} of the 192-wide gate
// GEMMs -> GRU update fully register-resident. Weights pinned as VGPR B-frags.
// xt gather software-pipelined one step ahead (loads in flight across the WMMAs).
__global__ __launch_bounds__(128) void rnn_kernel(
    const int*   __restrict__ item_seqs,   // (B,S)
    const float* __restrict__ emb,         // (V,E) f32
    const bf16*  __restrict__ Wih_bf,      // (192,64)
    const bf16*  __restrict__ Whh_bf,      // (192,64)
    const bf16*  __restrict__ Wp_bf,       // (64,64)
    const bf16*  __restrict__ Up_bf,       // (64,64)
    const bf16*  __restrict__ wqp_bf,      // (16,64) padded
    const bf16*  __restrict__ uqp_bf,      // (16,64) padded
    const float* __restrict__ Wp_b, const float* __restrict__ Up_b,
    const float* __restrict__ wq_b, const float* __restrict__ uq_b,
    const float* __restrict__ b_ih, const float* __restrict__ b_hh,
    const float* __restrict__ gumbel,      // (P,S,B,2)
    float*       __restrict__ out_gates,   // (B,P,S)
    float*       __restrict__ ps_out)      // (B,64) f32
{
    __shared__ bf16  xt_s[2][1024];
    __shared__ float xt_f[2][1024];
    __shared__ bf16  h_s[1024];
    __shared__ bf16  xpg_s[1024];
    __shared__ bf16  ps_bf[1024];
    __shared__ float logit_s[32];
    __shared__ float bg_s[16];
    __shared__ int   idx_s[2][16];

    const int tid  = threadIdx.x;
    const int lane = tid & 31;
    const int wid  = tid >> 5;
    const int half = lane >> 4;
    const int n15  = lane & 15;
    const int cw   = wid * 16 + n15;      // this lane's output column
    const int b0   = blockIdx.x * 16;

    // ---- pin weight B-fragments in registers ----
    v16bf bWp[2], bIh[3][2], bHh[3][2];
#pragma unroll
    for (int kk = 0; kk < 2; ++kk) bWp[kk] = bfrag(Wp_bf + wid * 1024, lane, kk);
#pragma unroll
    for (int j = 0; j < 3; ++j)
#pragma unroll
        for (int kk = 0; kk < 2; ++kk) {
            bIh[j][kk] = bfrag(Wih_bf + (wid + j * 4) * 1024, lane, kk);
            bHh[j][kk] = bfrag(Whh_bf + (wid + j * 4) * 1024, lane, kk);
        }

    // per-lane bias scalars (column cw of each gate third)
    const float wpbR = Wp_b[cw];
    const float upbR = Up_b[cw];
    const float bihR = b_ih[cw], bihZ = b_ih[64 + cw], bihN = b_ih[128 + cw];
    const float bhhR = b_hh[cw], bhhZ = b_hh[64 + cw], bhhN = b_hh[128 + cw];
    const float qb0 = wq_b[0] + uq_b[0];
    const float qb1 = wq_b[1] + uq_b[1];

    float psF[8], hF[8], ppA[8];
#pragma unroll
    for (int r = 0; r < 8; ++r) psF[r] = 0.0f;

    for (int p = 0; p < PPASS; ++p) {
        // ---- pass prologue: stage pre_prefer, reset h, load idx[0..1] ----
#pragma unroll
        for (int r = 0; r < 8; ++r) {
            int m = r + (half << 3);
            ps_bf[m * 64 + cw] = (bf16)psF[r];
            h_s[m * 64 + cw]   = (bf16)0.0f;
            hF[r] = 0.0f;
        }
        if (tid < 16)                idx_s[0][tid]      = item_seqs[(b0 + tid) * SSQ + 0];
        else if (tid < 32)           idx_s[1][tid - 16] = item_seqs[(b0 + tid - 16) * SSQ + 1];
        __syncthreads();
        {   // ppaff = pre_prefer @ Up^T + Up_b  (constant over the pass)
            v8f acc = V8F_ZERO;
#pragma unroll
            for (int kk = 0; kk < 2; ++kk)
                acc = wmma_bf(afrag(ps_bf, lane, kk),
                              bfrag(Up_bf + wid * 1024, lane, kk), acc);
#pragma unroll
            for (int r = 0; r < 8; ++r) ppA[r] = acc[r] + upbR;
        }
        // gather xt for s=0 into buffer 0 (same phase as ppaff GEMM)
        for (int i = tid; i < 1024; i += 128) {
            int m = i >> 6, c = i & 63;
            float v = emb[(size_t)idx_s[0][m] * 64 + c];
            xt_f[0][i] = v;
            xt_s[0][i] = (bf16)v;
        }
        __syncthreads();

        for (int s = 0; s < SSQ; ++s) {
            const int cur = s & 1;
            // ---- phase A: issue next-step gather (stays in flight), prefer gate ----
            float pf[8];
            if (s + 1 < SSQ) {
#pragma unroll
                for (int k = 0; k < 8; ++k) {
                    int i = tid + 128 * k;
                    int m = i >> 6, c = i & 63;
                    pf[k] = emb[(size_t)idx_s[(s + 1) & 1][m] * 64 + c];
                }
            }
            {
                v8f acc = V8F_ZERO;
#pragma unroll
                for (int kk = 0; kk < 2; ++kk)
                    acc = wmma_bf(afrag(xt_s[cur], lane, kk), bWp[kk], acc);
#pragma unroll
                for (int r = 0; r < 8; ++r) {
                    int m = r + (half << 3);
                    float pg = sigmf(acc[r] + wpbR + ppA[r]);
                    xpg_s[m * 64 + cw] = (bf16)(xt_f[cur][m * 64 + cw] * pg);
                }
            }
            __syncthreads();
            // ---- phase B: gate GEMMs into registers; wave0 leap gate; wave1 idx+2 ----
            v8f giR = V8F_ZERO, giZ = V8F_ZERO, giN = V8F_ZERO;
            v8f ghR = V8F_ZERO, ghZ = V8F_ZERO, ghN = V8F_ZERO;
            {
                v16bf x0 = afrag(xt_s[cur], lane, 0), x1 = afrag(xt_s[cur], lane, 1);
                giR = wmma_bf(x0, bIh[0][0], giR); giR = wmma_bf(x1, bIh[0][1], giR);
                giZ = wmma_bf(x0, bIh[1][0], giZ); giZ = wmma_bf(x1, bIh[1][1], giZ);
                giN = wmma_bf(x0, bIh[2][0], giN); giN = wmma_bf(x1, bIh[2][1], giN);
                v16bf h0 = afrag(h_s, lane, 0), h1 = afrag(h_s, lane, 1);
                ghR = wmma_bf(h0, bHh[0][0], ghR); ghR = wmma_bf(h1, bHh[0][1], ghR);
                ghZ = wmma_bf(h0, bHh[1][0], ghZ); ghZ = wmma_bf(h1, bHh[1][1], ghZ);
                ghN = wmma_bf(h0, bHh[2][0], ghN); ghN = wmma_bf(h1, bHh[2][1], ghN);
            }
            if (wid == 0) {
                v8f lac = V8F_ZERO;
#pragma unroll
                for (int kk = 0; kk < 2; ++kk)
                    lac = wmma_bf(afrag(h_s, lane, kk),   bfrag(wqp_bf, lane, kk), lac);
#pragma unroll
                for (int kk = 0; kk < 2; ++kk)
                    lac = wmma_bf(afrag(xpg_s, lane, kk), bfrag(uqp_bf, lane, kk), lac);
                if (n15 < 2) {
#pragma unroll
                    for (int r = 0; r < 8; ++r) {
                        int m = r + (half << 3);
                        logit_s[m * 2 + n15] = lac[r] + (n15 == 0 ? qb0 : qb1);
                    }
                }
                // LDS ops are in-order within a wave: safe to read logits back now.
                if (lane < 16) {
                    size_t gb = (((size_t)p * SSQ + s) * NB + (size_t)(b0 + lane)) * 2;
                    float s0 = logit_s[lane * 2 + 0] + gumbel[gb + 0];
                    float s1 = logit_s[lane * 2 + 1] + gumbel[gb + 1];
                    float bg = (s0 >= s1) ? 1.0f : 0.0f; // argmax(log_softmax(l)+g)==argmax(l+g)
                    bg_s[lane] = bg;
                    out_gates[((size_t)(b0 + lane) * PPASS + p) * SSQ + s] = bg;
                }
            }
            if (wid == 1 && lane < 16 && (s + 2) < SSQ)
                idx_s[s & 1][lane] = item_seqs[(b0 + lane) * SSQ + (s + 2)];
            __syncthreads();
            // ---- phase C: register-resident GRU update + deposit prefetched xt ----
#pragma unroll
            for (int r = 0; r < 8; ++r) {
                int m = r + (half << 3);
                float gr = sigmf(giR[r] + bihR + ghR[r] + bhhR);
                float gz = sigmf(giZ[r] + bihZ + ghZ[r] + bhhZ);
                float gn = tanhf_fast(giN[r] + bihN + gr * (ghN[r] + bhhN));
                float hp = hF[r];
                float hg = (1.0f - gz) * gn + gz * hp;
                float bg = bg_s[m];
                float hn = hg * bg + hp * (1.0f - bg);
                hF[r] = hn;
                h_s[m * 64 + cw] = (bf16)hn;
            }
            if (s + 1 < SSQ) {
#pragma unroll
                for (int k = 0; k < 8; ++k) {
                    int i = tid + 128 * k;
                    xt_f[1 - cur][i] = pf[k];
                    xt_s[1 - cur][i] = (bf16)pf[k];
                }
            }
            __syncthreads();
        }
#pragma unroll
        for (int r = 0; r < 8; ++r) psF[r] += hF[r];
    }
#pragma unroll
    for (int r = 0; r < 8; ++r) {
        int m = r + (half << 3);
        ps_out[(size_t)(b0 + m) * 64 + cw] = psF[r];
    }
}

// ---------------- ideal = prefer_sum @ Ww^T + Wb  -> bf16 ----------------
__global__ __launch_bounds__(32) void ideal_kernel(const float* __restrict__ ps,
                                                   const bf16* __restrict__ Ww_bf,
                                                   const float* __restrict__ Wb,
                                                   bf16* __restrict__ ideal_bf) {
    __shared__ bf16 a_s[1024];
    const int lane = threadIdx.x;
    const int half = lane >> 4;
    const int b0   = blockIdx.x * 16;
    for (int i = lane; i < 1024; i += 32) a_s[i] = (bf16)ps[(size_t)b0 * 64 + i];
    __syncthreads();
#pragma unroll
    for (int nt = 0; nt < 4; ++nt) {
        v8f acc = V8F_ZERO;
#pragma unroll
        for (int kk = 0; kk < 2; ++kk)
            acc = wmma_bf(afrag(a_s, lane, kk), bfrag(Ww_bf + nt * 1024, lane, kk), acc);
        int n = nt * 16 + (lane & 15);
#pragma unroll
        for (int r = 0; r < 8; ++r) {
            int m = r + (half << 3);
            ideal_bf[(size_t)(b0 + m) * 64 + n] = (bf16)(acc[r] + Wb[n]);
        }
    }
}

// ---------------- scores = ideal @ emb^T  (B x V) ----------------
__global__ __launch_bounds__(32) void scores_kernel(const bf16* __restrict__ ideal_bf,
                                                    const bf16* __restrict__ emb_bf,
                                                    float* __restrict__ out) {
    const int lane = threadIdx.x;
    const int half = lane >> 4;
    const int b0   = blockIdx.y * 16;
    const int nt0  = blockIdx.x * 5;     // 3125 n-tiles = 625 blocks * 5
    const bf16* Abase = ideal_bf + (size_t)b0 * 64;
    v16bf a0 = afrag(Abase, lane, 0), a1 = afrag(Abase, lane, 1);
#pragma unroll
    for (int t = 0; t < 5; ++t) {
        int nt = nt0 + t;
        v8f acc = V8F_ZERO;
        acc = wmma_bf(a0, bfrag(emb_bf + (size_t)nt * 1024, lane, 0), acc);
        acc = wmma_bf(a1, bfrag(emb_bf + (size_t)nt * 1024, lane, 1), acc);
        int n = nt * 16 + (lane & 15);
#pragma unroll
        for (int r = 0; r < 8; ++r) {
            int m = r + (half << 3);
            out[(size_t)(b0 + m) * VV + n] = acc[r];
        }
    }
}

// ---------------- per-row logsumexp over V ----------------
__global__ __launch_bounds__(256) void lse_kernel(const float* __restrict__ scores,
                                                  float* __restrict__ lse) {
    __shared__ float red[256];
    const int r = blockIdx.x;
    const float* row = scores + (size_t)r * VV;
    float mx = -3.4e38f;
    for (int i = threadIdx.x; i < VV; i += 256) mx = fmaxf(mx, row[i]);
    red[threadIdx.x] = mx;
    __syncthreads();
    for (int o = 128; o > 0; o >>= 1) {
        if (threadIdx.x < o) red[threadIdx.x] = fmaxf(red[threadIdx.x], red[threadIdx.x + o]);
        __syncthreads();
    }
    mx = red[0];
    __syncthreads();
    float sm = 0.0f;
    for (int i = threadIdx.x; i < VV; i += 256) sm += __expf(row[i] - mx);
    red[threadIdx.x] = sm;
    __syncthreads();
    for (int o = 128; o > 0; o >>= 1) {
        if (threadIdx.x < o) red[threadIdx.x] += red[threadIdx.x + o];
        __syncthreads();
    }
    if (threadIdx.x == 0) lse[r] = mx + __logf(red[0]);
}

__global__ void norm_kernel(float* __restrict__ out, const float* __restrict__ lse) {
    const int r = blockIdx.y;
    const int i = blockIdx.x * blockDim.x + threadIdx.x;
    if (i < VV) out[(size_t)r * VV + i] -= lse[r];
}

extern "C" void kernel_launch(void* const* d_in, const int* in_sizes, int n_in,
                              void* d_out, int out_size, void* d_ws, size_t ws_size,
                              hipStream_t stream) {
    const int*   item_seqs = (const int*)d_in[0];
    const float* emb   = (const float*)d_in[1];
    const float* Wp_w  = (const float*)d_in[2];
    const float* Wp_b  = (const float*)d_in[3];
    const float* Up_w  = (const float*)d_in[4];
    const float* Up_b  = (const float*)d_in[5];
    const float* wq_w  = (const float*)d_in[6];
    const float* wq_b  = (const float*)d_in[7];
    const float* uq_w  = (const float*)d_in[8];
    const float* uq_b  = (const float*)d_in[9];
    const float* W_ih  = (const float*)d_in[10];
    const float* b_ih  = (const float*)d_in[11];
    const float* W_hh  = (const float*)d_in[12];
    const float* b_hh  = (const float*)d_in[13];
    const float* Ww    = (const float*)d_in[14];
    const float* Wb    = (const float*)d_in[15];
    const float* gumbel = (const float*)d_in[16];

    float* out_scores = (float*)d_out;                       // (B, V)
    float* out_gates  = (float*)d_out + (size_t)NB * VV;     // (B, P, S)

    char* ws = (char*)d_ws;
    bf16* emb_bf = (bf16*)(ws);       size_t o = (size_t)VV * 64 * 2;
    bf16* Wih_bf = (bf16*)(ws + o);   o += (size_t)G3 * 64 * 2;
    bf16* Whh_bf = (bf16*)(ws + o);   o += (size_t)G3 * 64 * 2;
    bf16* Wp_bf  = (bf16*)(ws + o);   o += (size_t)64 * 64 * 2;
    bf16* Up_bf  = (bf16*)(ws + o);   o += (size_t)64 * 64 * 2;
    bf16* Ww_bf  = (bf16*)(ws + o);   o += (size_t)64 * 64 * 2;
    bf16* wqp_bf = (bf16*)(ws + o);   o += (size_t)16 * 64 * 2;
    bf16* uqp_bf = (bf16*)(ws + o);   o += (size_t)16 * 64 * 2;
    float* ps_ws = (float*)(ws + o);  o += (size_t)NB * 64 * 4;
    bf16* ideal_bf = (bf16*)(ws + o); o += (size_t)NB * 64 * 2;
    float* lse_ws = (float*)(ws + o);

    {
        int n = VV * 64;
        cvt_bf16_kernel<<<(n + 255) / 256, 256, 0, stream>>>(emb, emb_bf, n);
        n = G3 * 64;
        cvt_bf16_kernel<<<(n + 255) / 256, 256, 0, stream>>>(W_ih, Wih_bf, n);
        cvt_bf16_kernel<<<(n + 255) / 256, 256, 0, stream>>>(W_hh, Whh_bf, n);
        n = 64 * 64;
        cvt_bf16_kernel<<<(n + 255) / 256, 256, 0, stream>>>(Wp_w, Wp_bf, n);
        cvt_bf16_kernel<<<(n + 255) / 256, 256, 0, stream>>>(Up_w, Up_bf, n);
        cvt_bf16_kernel<<<(n + 255) / 256, 256, 0, stream>>>(Ww, Ww_bf, n);
        padq_kernel<<<4, 256, 0, stream>>>(wq_w, uq_w, wqp_bf, uqp_bf);
    }
    rnn_kernel<<<NB / 16, 128, 0, stream>>>(item_seqs, emb,
                                            Wih_bf, Whh_bf, Wp_bf, Up_bf, wqp_bf, uqp_bf,
                                            Wp_b, Up_b, wq_b, uq_b, b_ih, b_hh,
                                            gumbel, out_gates, ps_ws);
    ideal_kernel<<<NB / 16, 32, 0, stream>>>(ps_ws, Ww_bf, Wb, ideal_bf);
    scores_kernel<<<dim3(625, NB / 16), 32, 0, stream>>>(ideal_bf, emb_bf, out_scores);
    lse_kernel<<<NB, 256, 0, stream>>>(out_scores, lse_ws);
    norm_kernel<<<dim3((VV + 255) / 256, NB), 256, 0, stream>>>(out_scores, lse_ws);
    (void)in_sizes; (void)n_in; (void)out_size; (void)ws_size;
}
